// MultiHeadAttention_84250078478522
// MI455X (gfx1250) — matrix operand order repaired and checked by reading, not compiled, not run on hardware
//
#include <hip/hip_runtime.h>
#include <hip/hip_bf16.h>

// MI455X / gfx1250: wave32, WMMA 16x16x32 f16 (f32 accum).
typedef __attribute__((ext_vector_type(16))) _Float16 v16h;
typedef __attribute__((ext_vector_type(8)))  _Float16 v8h;
typedef __attribute__((ext_vector_type(8)))  float    v8f;

#define BB   4
#define SS   2048
#define DD   1024      // model dim == H*DK == H*DV
#define HH   16
#define DHE  64        // head dim (DK == DV)

#define NEG_INF (-__builtin_inff())

// ---- WMMA operand layouts (CDNA5 ISA 7.12.2) ------------------------------
// A (f16, 16x32 MxK): lane (m = l&15, hb = l>>4):
//     halves 0..7  = K[hb*8 .. hb*8+7], halves 8..15 = K[16+hb*8 ..]  (2x16B)
// B (f16, 32x16 KxN): lane = N + 16*hb, halves in the same K order as A
// C/D (f32 16x16):    lane = N + 16*(M>>3), vgpr = M&7

__device__ __forceinline__ v16h load_a16(const _Float16* p0, const _Float16* p1) {
  v8h a = *(const v8h*)p0;
  v8h b = *(const v8h*)p1;
  v16h r;
#pragma unroll
  for (int i = 0; i < 8; ++i) { r[i] = a[i]; r[8 + i] = b[i]; }
  return r;
}

__device__ __forceinline__ v8f wmma_f16(v16h a, v16h b, v8f c) {
  return __builtin_amdgcn_wmma_f32_16x16x32_f16(false, a, false, b,
                                                (short)0, c, false, false);
}

// ---------------------------------------------------------------------------
// Prep 1: X fp32 -> f16 row-major (8 elements / thread, 2x b128 in / b128 out)
// ---------------------------------------------------------------------------
__global__ __launch_bounds__(256)
void cvt_x_kernel(const float* __restrict__ X, _Float16* __restrict__ Xh)
{
  const size_t i = ((size_t)blockIdx.x * 256 + threadIdx.x) * 8;
  v8f x = *(const v8f*)(X + i);
  v8h o;
#pragma unroll
  for (int j = 0; j < 8; ++j) o[j] = (_Float16)x[j];
  *(v8h*)(Xh + i) = o;
}

// ---------------------------------------------------------------------------
// Prep 2: W [K=1024][N=1024] fp32 -> Wt [N][K] f16, 32x32 LDS tile transpose
// ---------------------------------------------------------------------------
__global__ __launch_bounds__(256)
void transpose_w_kernel(const float* __restrict__ W, _Float16* __restrict__ Wt)
{
  __shared__ _Float16 tile[32][33];           // padded to dodge bank conflicts
  const int n0 = blockIdx.x * 32;
  const int k0 = blockIdx.y * 32;
  const int tx = threadIdx.x;                 // 0..31
  const int ty = threadIdx.y;                 // 0..7
#pragma unroll
  for (int i = 0; i < 32; i += 8)
    tile[ty + i][tx] = (_Float16)W[(size_t)(k0 + ty + i) * DD + n0 + tx];
  __syncthreads();
#pragma unroll
  for (int i = 0; i < 32; i += 8)
    Wt[(size_t)(n0 + ty + i) * DD + k0 + tx] = tile[tx][ty + i];
}

// ---------------------------------------------------------------------------
// GEMM: dst = A(f16 [8192][1024]) @ Wt^T + bias, one wave -> 32x64 tile.
// mode 0: f16 store [B,H,S,64]    (Q, K)
// mode 1: f16 store [B,H,64,S]    (V transposed)
// mode 2: f32 store [B*S, 1024]   (final output)
// ---------------------------------------------------------------------------
__global__ __launch_bounds__(32)
void gemm_kernel(const _Float16* __restrict__ A, const _Float16* __restrict__ Wt,
                 const float* __restrict__ bias, void* __restrict__ dst, int mode)
{
  const int lane = threadIdx.x;
  const int nloc = lane & 15;
  const int hb   = lane >> 4;
  const int row0 = blockIdx.x * 32;           // flat row over B*S
  const int col0 = blockIdx.y * 64;

  v8f acc[2][4];
#pragma unroll
  for (int u = 0; u < 2; ++u)
#pragma unroll
    for (int t = 0; t < 4; ++t)
#pragma unroll
      for (int i = 0; i < 8; ++i) acc[u][t][i] = 0.0f;

  for (int k0 = 0; k0 < DD; k0 += 32) {
    v16h a[2];
#pragma unroll
    for (int u = 0; u < 2; ++u) {
      const _Float16* ap = A + (size_t)(row0 + u * 16 + nloc) * DD + k0 + hb * 8;
      a[u] = load_a16(ap, ap + 16);
      __builtin_prefetch(ap + 32, 0, 1);      // next k-slab (global_prefetch_b8)
    }
#pragma unroll
    for (int t = 0; t < 4; ++t) {
      const _Float16* wp = Wt + (size_t)(col0 + t * 16 + nloc) * DD + k0 + hb * 8;
      v16h bf = load_a16(wp, wp + 16);
      __builtin_prefetch(wp + 32, 0, 1);
      acc[0][t] = wmma_f16(a[0], bf, acc[0][t]);
      acc[1][t] = wmma_f16(a[1], bf, acc[1][t]);
    }
  }

#pragma unroll
  for (int u = 0; u < 2; ++u) {
    const int row = row0 + u * 16;
    const int b   = row >> 11;                // / S
    const int s0  = row & (SS - 1);
#pragma unroll
    for (int t = 0; t < 4; ++t) {
      const int col = col0 + t * 16 + nloc;
      const float bv = bias[col];
      const int head = col >> 6;
      const int dk   = col & 63;
      if (mode == 0) {
        _Float16* q = (_Float16*)dst +
                      (((size_t)b * HH + head) * SS + s0) * DHE + dk;
#pragma unroll
        for (int r = 0; r < 8; ++r)
          q[(size_t)(r + 8 * hb) * DHE] = (_Float16)(acc[u][t][r] + bv);
      } else if (mode == 1) {
        v8h pk;
#pragma unroll
        for (int r = 0; r < 8; ++r) pk[r] = (_Float16)(acc[u][t][r] + bv);
        _Float16* q = (_Float16*)dst +
                      (((size_t)b * HH + head) * DHE + dk) * SS + s0 + 8 * hb;
        *(v8h*)q = pk;
      } else {
        float* y = (float*)dst + (size_t)row * DD + col;
#pragma unroll
        for (int r = 0; r < 8; ++r)
          y[(size_t)(r + 8 * hb) * DD] = acc[u][t][r] + bv;
      }
    }
  }
}

// ---------------------------------------------------------------------------
// Causal flash attention, one wave per (b, h, 16-query tile).
// scores^T = K_tile(16x64) x Q^T  -> softmax rows are lane-local
// (8 regs + one shfl_xor 16).  O^T += V^T_tile(16x32) x P^T(32x16).
// ---------------------------------------------------------------------------
__global__ __launch_bounds__(32)
void attn_kernel(const _Float16* __restrict__ Qf, const _Float16* __restrict__ Kf,
                 const _Float16* __restrict__ Vt, _Float16* __restrict__ Of)
{
  const int lane = threadIdx.x;
  const int nloc = lane & 15;   // query within tile (N of score/out tiles)
  const int hb   = lane >> 4;
  const int q0   = blockIdx.x * 16;
  const int h    = blockIdx.y;
  const int b    = blockIdx.z;

  const _Float16* Qp = Qf + (((size_t)b * HH + h) * SS) * DHE;
  const _Float16* Kp = Kf + (((size_t)b * HH + h) * SS) * DHE;
  const _Float16* Vp = Vt + (((size_t)b * HH + h) * DHE) * SS;
  _Float16*       Op = Of + ((size_t)b * SS + q0) * (HH * DHE) + h * DHE;

  // Q rows as B-fragments for the two 32-wide d-steps (2x16B/lane, same k order as A)
  v16h bq[2];
#pragma unroll
  for (int kk = 0; kk < 2; ++kk) {
    const _Float16* qp = Qp + (size_t)(q0 + nloc) * DHE + kk * 32 + hb * 8;
    bq[kk] = load_a16(qp, qp + 16);
  }

  v8f oacc[4];
#pragma unroll
  for (int t = 0; t < 4; ++t)
#pragma unroll
    for (int i = 0; i < 8; ++i) oacc[t][i] = 0.0f;
  float mrun = NEG_INF;
  float lrun = 0.0f;

  const int kb_last = ((q0 + 15) >> 5) << 5;
  for (int kb = 0; kb <= kb_last; kb += 32) {
    // ---- scores^T for 32 keys: two 16-key M-tiles, two d-steps each ----
    v8f s[2];
#pragma unroll
    for (int tt = 0; tt < 2; ++tt) {
      const _Float16* kr = Kp + (size_t)(kb + tt * 16 + nloc) * DHE;
      v16h a0 = load_a16(kr + hb * 8,      kr + 16 + hb * 8);
      v16h a1 = load_a16(kr + 32 + hb * 8, kr + 48 + hb * 8);
      v8f z;
#pragma unroll
      for (int i = 0; i < 8; ++i) z[i] = 0.0f;
      z = wmma_f16(a0, bq[0], z);
      z = wmma_f16(a1, bq[1], z);
      s[tt] = z;
    }
    // ---- scale, causal mask, online-softmax stats ----
    float mloc = NEG_INF;
#pragma unroll
    for (int tt = 0; tt < 2; ++tt)
#pragma unroll
      for (int r = 0; r < 8; ++r) {
        const int key = kb + tt * 16 + r + 8 * hb;       // D-layout: M = r+8*hb
        float v = s[tt][r] * 0.125f;                     // 1/sqrt(64)
        if (key > q0 + nloc) v = NEG_INF;                // causal
        s[tt][r] = v;
        mloc = fmaxf(mloc, v);
      }
    const float mnew  = fmaxf(mrun, fmaxf(mloc, __shfl_xor(mloc, 16, 32)));
    const float alpha = __expf(mrun - mnew);
    float p0[8], p1[8], ls = 0.0f;
#pragma unroll
    for (int r = 0; r < 8; ++r) { p0[r] = __expf(s[0][r] - mnew); ls += p0[r]; }
#pragma unroll
    for (int r = 0; r < 8; ++r) { p1[r] = __expf(s[1][r] - mnew); ls += p1[r]; }
    lrun = lrun * alpha + (ls + __shfl_xor(ls, 16, 32));
    mrun = mnew;
#pragma unroll
    for (int t = 0; t < 4; ++t)
#pragma unroll
      for (int i = 0; i < 8; ++i) oacc[t][i] *= alpha;   // per-query rescale

    // ---- repack P^T into B-operand layout (one cross-half swap) ----
    // B half h on lane-half hb corresponds to local key = hb*16 + h.
    v16h pfrag;
#pragma unroll
    for (int r = 0; r < 8; ++r) {
      const float q0s = __shfl_xor(p0[r], 16, 32);
      const float q1s = __shfl_xor(p1[r], 16, 32);
      pfrag[r]     = (_Float16)(hb ? q1s  : p0[r]);      // keys hb*16 + 0..7
      pfrag[8 + r] = (_Float16)(hb ? p1[r] : q0s);       // keys hb*16 + 8..15
    }
    // ---- O^T += V^T x P^T ----
#pragma unroll
    for (int t = 0; t < 4; ++t) {
      const _Float16* vr = Vp + (size_t)(t * 16 + nloc) * SS + kb;
      v16h av = *(const v16h*)(vr + hb * 16);
      oacc[t] = wmma_f16(av, pfrag, oacc[t]);
    }
  }

  const float inv = 1.0f / lrun;   // identical in both lane halves
#pragma unroll
  for (int t = 0; t < 4; ++t) {
    v8h pk;
#pragma unroll
    for (int r = 0; r < 8; ++r) pk[r] = (_Float16)(oacc[t][r] * inv);
    // O^T tile element (M=vcol, N=query): contiguous along vcol per lane
    _Float16* op = Op + (size_t)nloc * (HH * DHE) + t * 16 + 8 * hb;
    *(v8h*)op = pk;
  }
}

// ---------------------------------------------------------------------------
extern "C" void kernel_launch(void* const* d_in, const int* in_sizes, int n_in,
                              void* d_out, int out_size, void* d_ws, size_t ws_size,
                              hipStream_t stream) {
  const float* X   = (const float*)d_in[0];
  const float* Wq  = (const float*)d_in[1];
  const float* bq_ = (const float*)d_in[2];
  const float* Wk  = (const float*)d_in[3];
  const float* bk_ = (const float*)d_in[4];
  const float* Wv  = (const float*)d_in[5];
  const float* bv_ = (const float*)d_in[6];
  const float* Wo  = (const float*)d_in[7];
  const float* bo_ = (const float*)d_in[8];
  float* out = (float*)d_out;

  const size_t seg  = (size_t)BB * HH * SS * DHE;  // 8M f16 = 16 MB
  const size_t wseg = (size_t)DD * DD;             // 1M f16 = 2 MB
  _Float16* Xh  = (_Float16*)d_ws;                 // [B*S, D]    f16
  _Float16* Qf  = Xh  + seg;                       // [B,H,S,64]
  _Float16* Kf  = Qf  + seg;                       // [B,H,S,64]
  _Float16* Vt  = Kf  + seg;                       // [B,H,64,S]
  _Float16* Of  = Vt  + seg;                       // [B,S,H*64]
  _Float16* Wqt = Of  + seg;                       // [N,K] f16 transposed
  _Float16* Wkt = Wqt + wseg;
  _Float16* Wvt = Wkt + wseg;
  _Float16* Wot = Wvt + wseg;

  // --- one-time precision/layout prep (L2-resident afterwards) ---
  cvt_x_kernel<<<dim3((unsigned)(seg / (256 * 8))), dim3(256), 0, stream>>>(X, Xh);
  const dim3 tB(32, 8, 1), tG(DD / 32, DD / 32, 1);
  transpose_w_kernel<<<tG, tB, 0, stream>>>(Wq, Wqt);
  transpose_w_kernel<<<tG, tB, 0, stream>>>(Wk, Wkt);
  transpose_w_kernel<<<tG, tB, 0, stream>>>(Wv, Wvt);
  transpose_w_kernel<<<tG, tB, 0, stream>>>(Wo, Wot);

  // --- projections ---
  const dim3 blk(32, 1, 1);
  const dim3 gGemm((BB * SS) / 32, DD / 64, 1);    // 256 x 16 waves
  gemm_kernel<<<gGemm, blk, 0, stream>>>(Xh, Wqt, bq_, Qf, 0);
  gemm_kernel<<<gGemm, blk, 0, stream>>>(Xh, Wkt, bk_, Kf, 0);
  gemm_kernel<<<gGemm, blk, 0, stream>>>(Xh, Wvt, bv_, Vt, 1);

  // --- attention ---
  const dim3 gAttn(SS / 16, HH, BB);               // 128 x 16 x 4 waves
  attn_kernel<<<gAttn, blk, 0, stream>>>(Qf, Kf, Vt, Of);

  // --- output projection ---
  gemm_kernel<<<gGemm, blk, 0, stream>>>(Of, Wot, bo_, out, 2);
}
